// DCN2_65180423685096
// MI455X (gfx1250) — matrix operand order repaired and checked by reading, not compiled, not run on hardware
//
#include <hip/hip_runtime.h>
#include <hip/hip_bf16.h>

// ---------------------------------------------------------------------------
// DCNv1 forward for MI455X (gfx1250, wave32, WMMA, async LDS copies).
//   Stage 1: f16 im2col of x -> patches [M=32768][K=2304]   (plain 3x3)
//   Stage 2: offset GEMM  off[M,72] = patches x w_off^T (+bias), WMMA f16
//   Stage 3: deformable bilinear sampling OVERWRITES patches in place
//   Stage 4: main GEMM    out[M,256] = patches x w_conv^T,  WMMA f16
// Both GEMMs: double-buffered LDS A tile filled with
// global_load_async_to_lds_b128 (ASYNCcnt), B fragments straight from global
// (16-bit B layout: lane half h holds k = 8h..8h+7 and 16+8h..16+8h+7).
// ---------------------------------------------------------------------------

typedef __attribute__((ext_vector_type(16))) _Float16 v16h;
typedef __attribute__((ext_vector_type(8)))  float    v8f;

constexpr int Bn   = 8;
constexpr int Cin  = 256;
constexpr int Cout = 256;
constexpr int Hh   = 64;
constexpr int Ww   = 64;
constexpr int HW   = Hh * Ww;            // 4096
constexpr int DGn  = 4;
constexpr int OffC = DGn * 3 * 3 * 2;    // 72 offset channels
constexpr int OffN = 80;                 // padded to WMMA-friendly N
constexpr int Kdim = Cin * 9;            // 2304 (GEMM K)
constexpr int Mdim = Bn * HW;            // 32768 (GEMM M)
constexpr int KT   = 32;                 // K step per WMMA
constexpr int NT_K = Kdim / KT;          // 72 K-iterations

#define LDA 40   // padded halves per LDS A-tile row (80 B -> bank-spread)

// --------------------------- CDNA5 async helpers ---------------------------
__device__ __forceinline__ void dcn_async_b128(unsigned lds_off, const void* g)
{
  // ASYNCcnt-tracked direct global->LDS copy (no VGPR round trip).
  asm volatile("global_load_async_to_lds_b128 %0, %1, off"
               :: "v"(lds_off), "v"(g) : "memory");
}
__device__ __forceinline__ void dcn_async_wait0()
{
  asm volatile("s_wait_asynccnt 0x0" ::: "memory");
}

// ---------------------------------------------------------------------------
// Weight converts. Main: w_conv [Cout][Cin*9] -> f16 (k-contiguous B rows).
// Offset: w_off [72][Cin*9] -> f16 padded to [80][Cin*9] (rows 72..79 = 0).
// ---------------------------------------------------------------------------
__global__ __launch_bounds__(256) void dcn_convert_w(
    const float* __restrict__ w, _Float16* __restrict__ w16)
{
  int id = blockIdx.x * 256 + threadIdx.x;   // < Cout*Kdim
  w16[id] = (_Float16)w[id];
}

__global__ __launch_bounds__(256) void dcn_convert_woff(
    const float* __restrict__ w, _Float16* __restrict__ w16)
{
  int id = blockIdx.x * 256 + threadIdx.x;   // < OffN*Kdim = 184320
  w16[id] = (id < OffC * Kdim) ? (_Float16)w[id] : (_Float16)0.0f;
}

// ---------------------------------------------------------------------------
// Stage 1: plain f16 im2col of x (zero-padded 3x3 shifts) -> patches [m][k].
// One block per pixel m, one thread per channel c; a wave writes 288
// contiguous halves over the 9 kk steps.
// ---------------------------------------------------------------------------
__global__ __launch_bounds__(256) void dcn_im2col(
    const float* __restrict__ x, _Float16* __restrict__ patches)
{
  int m = blockIdx.x;
  int c = threadIdx.x;
  int b = m >> 12;
  int p = m & (HW - 1);
  int yi = p >> 6;
  int xi = p & (Ww - 1);
  const float* xpl = x + ((size_t)b * Cin + c) * HW;
  _Float16* dst = patches + (size_t)m * Kdim + c * 9;
#pragma unroll
  for (int kk = 0; kk < 9; ++kk) {
    int yy = yi + kk / 3 - 1;
    int xx = xi + kk % 3 - 1;
    float v = (yy >= 0 && yy < Hh && xx >= 0 && xx < Ww) ? xpl[yy * Ww + xx]
                                                         : 0.0f;
    dst[kk] = (_Float16)v;
  }
}

// ---------------------------------------------------------------------------
// Stage 2: offset GEMM. off[M, 72] = patches[M,K] x w_off16[80,K]^T + bias.
// Block: 256 thr = 8 waves, tile 128(M) x 80(N); wave = 16 rows x 5 N-frags.
// ---------------------------------------------------------------------------
__global__ __launch_bounds__(256) void dcn_gemm_off(
    const _Float16* __restrict__ Apat, const _Float16* __restrict__ Bw,
    const float* __restrict__ b_off, float* __restrict__ offbuf)
{
  __shared__ _Float16 lA[2 * 128 * LDA];

  const int tid  = threadIdx.x;
  const int lane = tid & 31;
  const int wave = tid >> 5;            // 0..7 -> M sub-tile
  const int m0   = blockIdx.x * 128;
  const int mw   = wave * 16;

  v8f acc[5];
#pragma unroll
  for (int nf = 0; nf < 5; ++nf) acc[nf] = (v8f){0,0,0,0,0,0,0,0};

  // cooperative A staging: each thread owns one 32 B row segment
  const int arow = tid >> 1;
  const int aseg = (tid & 1) << 4;      // halves
  const _Float16* agsrc = Apat + (size_t)(m0 + arow) * Kdim + aseg;
  const unsigned lrow = (unsigned)(uintptr_t)(lA + arow * LDA + aseg);
  const unsigned lstep = 128 * LDA * sizeof(_Float16);

  const int fl  = lane & 15;
  const int akb = (lane < 16) ? 0 : 8;
  const int bkb = (lane < 16) ? 0 : 8;

  // prologue: tile 0 -> buffer 0
  dcn_async_b128(lrow,      agsrc);
  dcn_async_b128(lrow + 16, agsrc + 8);

  for (int it = 0; it < NT_K; ++it) {
    const int k0 = it * KT;
    dcn_async_wait0();
    __syncthreads();
    if (it + 1 < NT_K) {
      const unsigned ld = lrow + ((it + 1) & 1) * lstep;
      dcn_async_b128(ld,      agsrc + k0 + KT);
      dcn_async_b128(ld + 16, agsrc + k0 + KT + 8);
    }

    const _Float16* base = lA + (it & 1) * (128 * LDA);
    v16h afr;
    {
      const _Float16* src = base + (mw + fl) * LDA + akb;
      ((uint4*)&afr)[0] = *(const uint4*)(src);
      ((uint4*)&afr)[1] = *(const uint4*)(src + 16);
    }
#pragma unroll
    for (int nf = 0; nf < 5; ++nf) {
      const _Float16* bsrc = Bw + (size_t)(nf * 16 + fl) * Kdim + k0 + bkb;
      v16h bfr;
      ((uint4*)&bfr)[0] = *(const uint4*)(bsrc);
      ((uint4*)&bfr)[1] = *(const uint4*)(bsrc + 16);
      acc[nf] = __builtin_amdgcn_wmma_f32_16x16x32_f16(
          false, afr, false, bfr, (short)0, acc[nf], false, false);
    }
  }

  // epilogue: C layout -> M = r + (lane>=16 ? 8 : 0), N = lane&15
  const int hi = lane >> 4;
  const int nn = lane & 15;
#pragma unroll
  for (int nf = 0; nf < 5; ++nf) {
    int n = nf * 16 + nn;
    if (n < OffC) {
      float bias = b_off[n];
#pragma unroll
      for (int r = 0; r < 8; ++r) {
        int m = m0 + mw + hi * 8 + r;
        int bb = m >> 12;
        int pp = m & (HW - 1);
        offbuf[((size_t)bb * OffC + n) * HW + pp] = acc[nf][r] + bias;
      }
    }
  }
}

// ---------------------------------------------------------------------------
// Stage 3: deformable bilinear sampling, overwrites patches in place.
// ---------------------------------------------------------------------------
__device__ __forceinline__ float dcn_tap(const float* __restrict__ pl,
                                         int yy, int xx)
{
  return (yy >= 0 && yy < Hh && xx >= 0 && xx < Ww) ? pl[yy * Ww + xx] : 0.0f;
}

__global__ __launch_bounds__(256) void dcn_deform(
    const float* __restrict__ x, const float* __restrict__ offbuf,
    _Float16* __restrict__ patches)
{
  int m = blockIdx.x;
  int c = threadIdx.x;
  int b = m >> 12;
  int p = m & (HW - 1);
  int yi = p >> 6;
  int xi = p & (Ww - 1);
  int dg = c >> 6;                       // Cg = 64

  const float* offs = offbuf + ((size_t)b * OffC + dg * 18) * HW + p;
  const float* xpl  = x + ((size_t)b * Cin + c) * HW;
  _Float16* dst = patches + (size_t)m * Kdim + c * 9;

#pragma unroll
  for (int kk = 0; kk < 9; ++kk) {
    int i = kk / 3, j = kk % 3;
    float oy = offs[(size_t)(kk * 2)     * HW];
    float ox = offs[(size_t)(kk * 2 + 1) * HW];
    float py = (float)(yi - 1 + i) + oy;
    float px = (float)(xi - 1 + j) + ox;
    float fy = floorf(py), fx = floorf(px);
    int   y0 = (int)fy,    x0 = (int)fx;
    float wy1 = py - fy, wy0 = 1.0f - wy1;
    float wx1 = px - fx, wx0 = 1.0f - wx1;
    float v = wy0 * wx0 * dcn_tap(xpl, y0,     x0)
            + wy0 * wx1 * dcn_tap(xpl, y0,     x0 + 1)
            + wy1 * wx0 * dcn_tap(xpl, y0 + 1, x0)
            + wy1 * wx1 * dcn_tap(xpl, y0 + 1, x0 + 1);
    dst[kk] = (_Float16)v;
  }
}

// ---------------------------------------------------------------------------
// Stage 4: main GEMM. Block 256 thr = 8 waves (4M x 2N), tile 128x128;
// wave = 32(M) x 64(N) = 2x4 WMMA accumulators.
// ---------------------------------------------------------------------------
__global__ __launch_bounds__(256) void dcn_gemm(
    const _Float16* __restrict__ Apat, const _Float16* __restrict__ Bw,
    float* __restrict__ out)
{
  __shared__ _Float16 lA[2 * 128 * LDA];

  const int tid  = threadIdx.x;
  const int lane = tid & 31;
  const int wave = tid >> 5;
  const int wm   = wave >> 1;           // 0..3
  const int wn   = wave & 1;            // 0..1

  const int m0 = blockIdx.x * 128;
  const int n0 = blockIdx.y * 128 + wn * 64;
  const int mw = wm * 32;

  v8f acc[2][4];
#pragma unroll
  for (int t = 0; t < 2; ++t)
#pragma unroll
    for (int nf = 0; nf < 4; ++nf) acc[t][nf] = (v8f){0,0,0,0,0,0,0,0};

  const int arow = tid >> 1;
  const int aseg = (tid & 1) << 4;
  const _Float16* agsrc = Apat + (size_t)(m0 + arow) * Kdim + aseg;
  const unsigned lrow = (unsigned)(uintptr_t)(lA + arow * LDA + aseg);
  const unsigned lstep = 128 * LDA * sizeof(_Float16);

  const int fl  = lane & 15;
  const int akb = (lane < 16) ? 0 : 8;
  const int bkb = (lane < 16) ? 0 : 8;

  dcn_async_b128(lrow,      agsrc);
  dcn_async_b128(lrow + 16, agsrc + 8);

  for (int it = 0; it < NT_K; ++it) {
    const int k0 = it * KT;
    dcn_async_wait0();
    __syncthreads();
    if (it + 1 < NT_K) {
      const unsigned ld = lrow + ((it + 1) & 1) * lstep;
      dcn_async_b128(ld,      agsrc + k0 + KT);
      dcn_async_b128(ld + 16, agsrc + k0 + KT + 8);
    }

    const _Float16* base = lA + (it & 1) * (128 * LDA);
    v16h afr[2];
#pragma unroll
    for (int t = 0; t < 2; ++t) {
      const _Float16* src = base + (mw + t * 16 + fl) * LDA + akb;
      ((uint4*)&afr[t])[0] = *(const uint4*)(src);
      ((uint4*)&afr[t])[1] = *(const uint4*)(src + 16);
    }
#pragma unroll
    for (int nf = 0; nf < 4; ++nf) {
      const _Float16* bsrc =
          Bw + (size_t)(n0 + nf * 16 + fl) * Kdim + k0 + bkb;
      v16h bfr;
      ((uint4*)&bfr)[0] = *(const uint4*)(bsrc);
      ((uint4*)&bfr)[1] = *(const uint4*)(bsrc + 16);
#pragma unroll
      for (int t = 0; t < 2; ++t) {
        acc[t][nf] = __builtin_amdgcn_wmma_f32_16x16x32_f16(
            false, afr[t], false, bfr, (short)0, acc[t][nf], false, false);
      }
    }
  }

  const int hi = lane >> 4;
  const int nn = lane & 15;
#pragma unroll
  for (int t = 0; t < 2; ++t) {
#pragma unroll
    for (int nf = 0; nf < 4; ++nf) {
#pragma unroll
      for (int r = 0; r < 8; ++r) {
        int m = m0 + mw + t * 16 + hi * 8 + r;
        int n = n0 + nf * 16 + nn;
        int bb = m >> 12;
        int pp = m & (HW - 1);
        out[((size_t)bb * Cout + n) * HW + pp] = acc[t][nf][r];
      }
    }
  }
}

// ---------------------------------------------------------------------------
// Launcher. Inputs: x[f32], w_off[f32], b_off[f32], w_conv[f32].
// ws: offbuf 9.44 MB | w16 1.18 MB | w16off 0.37 MB | patches 151 MB.
// patches is reused: plain im2col (stage 1/2) then deformed (stage 3/4).
// ---------------------------------------------------------------------------
extern "C" void kernel_launch(void* const* d_in, const int* in_sizes, int n_in,
                              void* d_out, int out_size, void* d_ws,
                              size_t ws_size, hipStream_t stream)
{
  (void)in_sizes; (void)n_in; (void)out_size; (void)ws_size;

  const float* x      = (const float*)d_in[0];
  const float* w_off  = (const float*)d_in[1];
  const float* b_off  = (const float*)d_in[2];
  const float* w_conv = (const float*)d_in[3];
  float* out = (float*)d_out;

  char* ws = (char*)d_ws;
  constexpr size_t OFF_BYTES  = (size_t)Bn * OffC * HW * sizeof(float);
  constexpr size_t W16_BYTES  = (size_t)Cout * Kdim * sizeof(_Float16);
  constexpr size_t WOFF_BYTES = (size_t)OffN * Kdim * sizeof(_Float16);
  float*    offbuf  = (float*)ws;
  _Float16* w16     = (_Float16*)(ws + OFF_BYTES);
  _Float16* w16off  = (_Float16*)(ws + OFF_BYTES + W16_BYTES);
  _Float16* patches = (_Float16*)(ws + OFF_BYTES + W16_BYTES + WOFF_BYTES);

  dcn_convert_w<<<(Cout * Kdim) / 256, 256, 0, stream>>>(w_conv, w16);
  dcn_convert_woff<<<(OffN * Kdim) / 256, 256, 0, stream>>>(w_off, w16off);

  // Stage 1: plain im2col -> patches
  dcn_im2col<<<Mdim, 256, 0, stream>>>(x, patches);

  // Stage 2: offset conv as WMMA GEMM (M=32768, N=80(72), K=2304)
  dcn_gemm_off<<<Mdim / 128, 256, 0, stream>>>(patches, w16off, b_off, offbuf);

  // Stage 3: deformable sampling, overwrite patches in place
  dcn_deform<<<Mdim, 256, 0, stream>>>(x, offbuf, patches);

  // Stage 4: main WMMA GEMM (M=32768, N=256, K=2304)
  dcn_gemm<<<dim3(Mdim / 128, Cout / 128), 256, 0, stream>>>(
      patches, w16, out);
}
